// LinearSelfAttnSeq_14714557956431
// MI455X (gfx1250) — matrix operand order repaired and checked by reading, not compiled, not run on hardware
//
#include <hip/hip_runtime.h>
#include <stdint.h>

// ---------------------------------------------------------------------------
// LinearSelfAttnSeq for MI455X (gfx1250, wave32, WMMA).
//  out = softmax((q@W^T + b) @ q^T / sqrt(D)) @ q
//  B=8, L=2048, D=512.  Compute-bound (~77 GFLOP vs ~68MB HBM traffic) ->
//  run all three matmuls on v_wmma_f32_16x16x32_bf16 with fp32 accumulation.
// Workspace layout (needs ~51 MB):
//   q_bf16  [B*L][D]   (A / B operand, row-major along d)
//   qp_bf16 [B*L][D]   (scaled projection, A operand of logits)
//   qT_bf16 [B][D][L]  (transposed keys/values, B operand of P@V)
//   W_bf16  [D][D]
// ---------------------------------------------------------------------------

typedef unsigned short u16;
typedef __attribute__((ext_vector_type(16))) __bf16 v16bf;
typedef __attribute__((ext_vector_type(8)))  float  v8f;

#define B_  8
#define L_  2048
#define D_  512
#define M_  (B_ * L_)   // 16384 total rows

union Frag { v16bf v; uint4 q[2]; };

__device__ __forceinline__ u16 f2bf(float f) {
  union { float f; uint32_t u; } c; c.f = f;
  const uint32_t u = c.u;
  return (u16)((u + 0x7FFFu + ((u >> 16) & 1u)) >> 16);   // round-nearest-even
}

// A-matrix fragment, 16x32 bf16 (M x K), src row-major with leading dim ld.
// ISA layout: lane l -> row (l&15); VGPR 0..3 hold K = h*8+0..7,
// VGPR 4..7 hold K = 16+h*8+0..7, h = l>>4.  => two b128 loads per lane.
__device__ __forceinline__ Frag load_a_frag(const u16* src, int ld, int row0, int k0) {
  const int lane = threadIdx.x & 31;
  const int m = lane & 15, h = lane >> 4;
  const u16* p = src + (size_t)(row0 + m) * ld + (k0 + h * 8);
  Frag f;
  f.q[0] = *(const uint4*)(p);
  f.q[1] = *(const uint4*)(p + 16);
  return f;
}

// B-matrix fragment, 32x16 bf16 (K x N), where B[k][n] = srcT[n0+n][k0+k]
// (srcT row-major along k with leading dim ld).  ISA layout: lane l -> col
// (l&15); VGPR 0..3 hold K = h*8+0..7, VGPR 4..7 hold K = 16+h*8+0..7.  => two b128 loads per lane.
__device__ __forceinline__ Frag load_b_frag(const u16* srcT, int ld, int n0, int k0) {
  const int lane = threadIdx.x & 31;
  const int n = lane & 15, h = lane >> 4;
  const u16* p = srcT + (size_t)(n0 + n) * ld + (k0 + h * 8);
  Frag f;
  f.q[0] = *(const uint4*)(p);
  f.q[1] = *(const uint4*)(p + 16);
  return f;
}

__device__ __forceinline__ v8f wmma_bf16(const Frag& a, const Frag& b, v8f c) {
  // (neg_a, A, neg_b, B, c_mod, C, reuse_a, reuse_b)
  return __builtin_amdgcn_wmma_f32_16x16x32_bf16(false, a.v, false, b.v,
                                                 (short)0, c, false, false);
}

// ---------------------------------------------------------------------------
// Stage 1a: fp32 -> bf16 elementwise
__global__ void cvt_kernel(const float* src, u16* dst, int n) {
  for (int i = blockIdx.x * blockDim.x + threadIdx.x; i < n;
       i += gridDim.x * blockDim.x)
    dst[i] = f2bf(src[i]);
}

// Stage 1b: qT[b][d][l] = bf16(q[b][l][d])  (32x32 tiles via LDS)
__global__ __launch_bounds__(256) void transpose_kernel(const float* q, u16* qT) {
  __shared__ u16 tile[32][33];
  const int bb = blockIdx.z;
  const int d0 = blockIdx.x * 32;
  const int l0 = blockIdx.y * 32;
  const int tx = threadIdx.x, ty = threadIdx.y;
  const float* src = q + ((size_t)bb * L_ + l0) * D_ + d0;
#pragma unroll
  for (int yy = 0; yy < 4; ++yy)
    tile[tx][ty + yy * 8] = f2bf(src[(size_t)(ty + yy * 8) * D_ + tx]);
  __syncthreads();
  u16* dst = qT + ((size_t)bb * D_ + d0) * L_ + l0;
#pragma unroll
  for (int yy = 0; yy < 4; ++yy)
    dst[(size_t)(ty + yy * 8) * L_ + tx] = tile[ty + yy * 8][tx];
}

// ---------------------------------------------------------------------------
// Stage 2: qp = bf16( (q @ W^T + bias) * 1/sqrt(D) )
// Wave tile 16x64 (4 C-frags); 8 waves -> WG tile 64x128.
__global__ __launch_bounds__(256) void proj_kernel(const u16* qb, const u16* Wb,
                                                   const float* bias, u16* qp) {
  const int wid  = threadIdx.x >> 5;
  const int lane = threadIdx.x & 31;
  const int nn = lane & 15, hh = lane >> 4;
  const int row0 = blockIdx.y * 64 + (wid >> 1) * 16;
  const int col0 = blockIdx.x * 128 + (wid & 1) * 64;

  v8f c[4];
#pragma unroll
  for (int j = 0; j < 4; ++j)
#pragma unroll
    for (int r = 0; r < 8; ++r) c[j][r] = 0.0f;

#pragma unroll 4
  for (int k0 = 0; k0 < D_; k0 += 32) {
    Frag a = load_a_frag(qb, D_, row0, k0);
#pragma unroll
    for (int j = 0; j < 4; ++j) {
      // B[k=d][n=e] = W[e][d]  -> srcT = W row-major
      Frag bf = load_b_frag(Wb, D_, col0 + j * 16, k0);
      c[j] = wmma_bf16(a, bf, c[j]);
    }
  }

  const float scale = 0.04419417382415922f;  // 1/sqrt(512), folded into qp
#pragma unroll
  for (int j = 0; j < 4; ++j) {
    const float bj = bias[col0 + j * 16 + nn];
#pragma unroll
    for (int r = 0; r < 8; ++r)
      qp[(size_t)(row0 + r + hh * 8) * D_ + col0 + j * 16 + nn] =
          f2bf((c[j][r] + bj) * scale);
  }
}

// ---------------------------------------------------------------------------
// Stage 3: flash attention.  One wave = 16 query rows, full online softmax,
// O accumulator = 32 f32 C-frags (16 x 512).  Key tile = 64.
__global__ __launch_bounds__(128) void attn_kernel(const u16* qp, const u16* qk,
                                                   const u16* vT, float* out) {
  __shared__ __align__(16) u16 Pbuf[4][16][64];   // per-wave P tile (bf16)
  const int wid  = threadIdx.x >> 5;
  const int lane = threadIdx.x & 31;
  const int nn = lane & 15, hh = lane >> 4;
  const int g   = blockIdx.x * 4 + wid;   // global wave id; 4 consecutive
  const int bb  = g >> 7;                 // waves share one batch -> WGP$ reuse
  const int qr0 = (g & 127) * 16;

  const u16* qpb = qp + (size_t)bb * L_ * D_;
  const u16* qkb = qk + (size_t)bb * L_ * D_;
  const u16* vTb = vT + (size_t)bb * D_ * L_;
  float* outb = out + (size_t)bb * L_ * D_;

  v8f o[D_ / 16];                         // 32 frags = 256 VGPRs/lane (ok, wave32 max 1024)
#pragma unroll
  for (int j = 0; j < D_ / 16; ++j)
#pragma unroll
    for (int r = 0; r < 8; ++r) o[j][r] = 0.0f;

  float mrow[8], lrow[8];                 // stats for rows (r + hh*8)
#pragma unroll
  for (int r = 0; r < 8; ++r) { mrow[r] = -3.0e38f; lrow[r] = 0.0f; }

  for (int n0 = 0; n0 < L_; n0 += 64) {
    // ---- S = qp_tile @ K_tile^T  (scale already folded into qp) ----
    v8f s[4];
#pragma unroll
    for (int j = 0; j < 4; ++j)
#pragma unroll
      for (int r = 0; r < 8; ++r) s[j][r] = 0.0f;

#pragma unroll 4
    for (int kt = 0; kt < D_; kt += 32) {
      Frag a = load_a_frag(qpb, D_, qr0, kt);
#pragma unroll
      for (int j = 0; j < 4; ++j) {
        // B[k=d][n=key] = q[key][d] -> srcT = q rows
        Frag bf = load_b_frag(qkb, D_, n0 + j * 16, kt);
        s[j] = wmma_bf16(a, bf, s[j]);
      }
    }

    // ---- online softmax (row = r + hh*8, cols striped over 16 lanes) ----
    float alpha[8], mnew[8];
#pragma unroll
    for (int r = 0; r < 8; ++r) {
      float v = fmaxf(fmaxf(s[0][r], s[1][r]), fmaxf(s[2][r], s[3][r]));
      v = fmaxf(v, __shfl_xor(v, 1, 32));
      v = fmaxf(v, __shfl_xor(v, 2, 32));
      v = fmaxf(v, __shfl_xor(v, 4, 32));
      v = fmaxf(v, __shfl_xor(v, 8, 32));
      mnew[r]  = fmaxf(mrow[r], v);
      alpha[r] = __expf(mrow[r] - mnew[r]);
      mrow[r]  = mnew[r];
    }
#pragma unroll
    for (int r = 0; r < 8; ++r) {
      float acc = 0.0f;
#pragma unroll
      for (int j = 0; j < 4; ++j) {
        const float p = __expf(s[j][r] - mnew[r]);
        s[j][r] = p;
        acc += p;
      }
      acc += __shfl_xor(acc, 1, 32);
      acc += __shfl_xor(acc, 2, 32);
      acc += __shfl_xor(acc, 4, 32);
      acc += __shfl_xor(acc, 8, 32);
      lrow[r] = lrow[r] * alpha[r] + acc;
    }

    // ---- P (C-layout f32) -> LDS bf16 row-major, then re-read as A-frag ----
#pragma unroll
    for (int j = 0; j < 4; ++j)
#pragma unroll
      for (int r = 0; r < 8; ++r)
        Pbuf[wid][r + hh * 8][j * 16 + nn] = f2bf(s[j][r]);

    // rescale O accumulators
#pragma unroll
    for (int j = 0; j < D_ / 16; ++j)
#pragma unroll
      for (int r = 0; r < 8; ++r) o[j][r] *= alpha[r];

    // ---- O += P @ V_tile  (V from transposed copy: contiguous B-frags) ----
#pragma unroll
    for (int h2 = 0; h2 < 2; ++h2) {
      Frag a = load_a_frag(&Pbuf[wid][0][0], 64, 0, h2 * 32);  // ds_load_b128
#pragma unroll
      for (int j = 0; j < D_ / 16; ++j) {
        // B[k=key][n=dcol] = v[key][dcol] = vT[dcol][key]
        Frag bf = load_b_frag(vTb, L_, j * 16, n0 + h2 * 32);
        o[j] = wmma_bf16(a, bf, o[j]);
      }
    }
  }

  // ---- epilogue: divide by softmax denominator, store fp32 ----
  float inv[8];
#pragma unroll
  for (int r = 0; r < 8; ++r) inv[r] = 1.0f / lrow[r];
#pragma unroll
  for (int j = 0; j < D_ / 16; ++j)
#pragma unroll
    for (int r = 0; r < 8; ++r)
      outb[(size_t)(qr0 + r + hh * 8) * D_ + j * 16 + nn] = o[j][r] * inv[r];
}

// ---------------------------------------------------------------------------
extern "C" void kernel_launch(void* const* d_in, const int* in_sizes, int n_in,
                              void* d_out, int out_size, void* d_ws, size_t ws_size,
                              hipStream_t stream) {
  (void)in_sizes; (void)n_in; (void)out_size; (void)ws_size;
  const float* q    = (const float*)d_in[0];
  const float* W    = (const float*)d_in[1];
  const float* bias = (const float*)d_in[2];
  float* out = (float*)d_out;

  u16* q_bf  = (u16*)d_ws;                       // 16 MiB
  u16* qp_bf = q_bf  + (size_t)M_ * D_;          // 16 MiB
  u16* qT_bf = qp_bf + (size_t)M_ * D_;          // 16 MiB (B*D*L == M*D)
  u16* W_bf  = qT_bf + (size_t)M_ * D_;          // 0.5 MiB   (total ~51 MB)

  cvt_kernel<<<4096, 256, 0, stream>>>(q, q_bf, M_ * D_);
  cvt_kernel<<<512, 256, 0, stream>>>(W, W_bf, D_ * D_);
  transpose_kernel<<<dim3(D_ / 32, L_ / 32, B_), dim3(32, 8), 0, stream>>>(q, qT_bf);
  proj_kernel<<<dim3(D_ / 128, M_ / 64), 256, 0, stream>>>(q_bf, W_bf, bias, qp_bf);
  attn_kernel<<<(B_ * (L_ / 16)) / 4, 128, 0, stream>>>(qp_bf, q_bf, qT_bf, out);
}
